// QuantAttention_27049704030389
// MI455X (gfx1250) — matrix operand order repaired and checked by reading, not compiled, not run on hardware
//
#include <hip/hip_runtime.h>
#include <hip/hip_bf16.h>

// ---------------------------------------------------------------------------
// QuantAttention forward for MI455X (gfx1250, wave32, WMMA f16 16x16x32)
// B=16 N=1024 C=768 H=12 D=64
// ---------------------------------------------------------------------------

#define Bv   16
#define Nv   1024
#define Cv   768
#define Hv   12
#define Dv   64
#define M1   (Bv * Nv)            // 16384
#define C3   (3 * Cv)             // 2304
#define OUT_ELEMS  ((long)Bv * Nv * Cv)               // 12,582,912
#define ATTN_ELEMS ((long)Bv * Hv * Nv * Nv)          // 201,326,592
#define HEAD_ELEMS ((long)Bv * Hv * Nv * Dv)          // 12,582,912

typedef __attribute__((ext_vector_type(16))) _Float16 v16h;
typedef __attribute__((ext_vector_type(8)))  float    v8f;
typedef int vi4 __attribute__((__vector_size__(4 * sizeof(int))));

struct QStat { unsigned encMin; unsigned encMax; float scale; float zp; };
// slots: 0=x 1=w_qkv 2=w_proj 3=q 4=k 5=v 6=attn 7=ctx

// Monotonic float<->uint encoding for atomic min/max on floats.
__device__ __forceinline__ unsigned encf(float f) {
  unsigned u = __float_as_uint(f);
  return (u >> 31) ? ~u : (u | 0x80000000u);
}
__device__ __forceinline__ float decf(unsigned e) {
  unsigned u = (e >> 31) ? (e & 0x7FFFFFFFu) : ~e;
  return __uint_as_float(u);
}

// fake_quant with precomputed reciprocal (avoid per-element f32 division).
__device__ __forceinline__ float fqv(float x, float s, float inv, float z) {
  float q = rintf(x * inv) + z;
  q = fminf(fmaxf(q, 0.0f), 255.0f);
  return (q - z) * s;
}

__device__ __forceinline__ unsigned packh2(float a, float b) {
  union { _Float16 h[2]; unsigned u; } t;
  t.h[0] = (_Float16)a; t.h[1] = (_Float16)b;
  return t.u;
}

// ---------------------------------------------------------------------------
// Stats kernels
// ---------------------------------------------------------------------------
__global__ void init_stats_kernel(QStat* st) {
  int i = threadIdx.x;
  if (i < 8) {
    st[i].encMin = 0xFFFFFFFFu;   // encodes +max -> good init for min
    st[i].encMax = 0u;            // encodes -max -> good init for max
    st[i].scale  = 1.0f;
    st[i].zp     = 0.0f;
    if (i == 6) st[i].encMin = 0x80000000u;  // attn >= 0 -> min starts at enc(+0)
  }
}

__global__ __launch_bounds__(256)
void reduce_minmax_kernel(const float* __restrict__ src, long nRows, int rowLen,
                          int rowStride, QStat* st) {
  long total = nRows * (long)rowLen;
  float lmin =  3.402823466e+38f;
  float lmax = -3.402823466e+38f;
  long stride = (long)gridDim.x * blockDim.x;
  for (long i = (long)blockIdx.x * blockDim.x + threadIdx.x; i < total; i += stride) {
    long r = i / rowLen;
    int  c = (int)(i - r * rowLen);
    float v = src[r * (long)rowStride + c];
    lmin = fminf(lmin, v);
    lmax = fmaxf(lmax, v);
  }
  __shared__ float smin[256], smax[256];
  int tid = threadIdx.x;
  smin[tid] = lmin; smax[tid] = lmax;
  __syncthreads();
  for (int s = 128; s > 0; s >>= 1) {
    if (tid < s) {
      smin[tid] = fminf(smin[tid], smin[tid + s]);
      smax[tid] = fmaxf(smax[tid], smax[tid + s]);
    }
    __syncthreads();
  }
  if (tid == 0) {
    atomicMin(&st->encMin, encf(smin[0]));
    atomicMax(&st->encMax, encf(smax[0]));
  }
}

__global__ void finalize_stats_kernel(QStat* st, unsigned mask) {
  int i = threadIdx.x;
  if (i >= 8 || !((mask >> i) & 1u)) return;
  float xmin = fminf(decf(st[i].encMin), 0.0f);
  float xmax = fmaxf(decf(st[i].encMax), 0.0f);
  float scale = fmaxf((xmax - xmin) / 255.0f, 1e-8f);
  st[i].scale = scale;
  st[i].zp    = rintf(-xmin / scale);
}

// ---------------------------------------------------------------------------
// Quantize / pack kernels (f32 -> fake-quantized f16)
// ---------------------------------------------------------------------------
__global__ __launch_bounds__(256)
void quantize_f16_kernel(const float* __restrict__ src, _Float16* __restrict__ dst,
                         long n, const QStat* __restrict__ st) {
  float s = st->scale, z = st->zp, inv = 1.0f / s;
  long stride = (long)gridDim.x * blockDim.x;
  for (long i = (long)blockIdx.x * blockDim.x + threadIdx.x; i < n; i += stride)
    dst[i] = (_Float16)fqv(src[i], s, inv, z);
}

// k16[((b*H+h)*N+n)*D+d] = fq(qkv[(b*N+n)*3C + C + h*D + d])
__global__ __launch_bounds__(256)
void pack_k_kernel(const float* __restrict__ qkv, _Float16* __restrict__ k16,
                   const QStat* __restrict__ st) {
  float s = st->scale, z = st->zp, inv = 1.0f / s;
  long stride = (long)gridDim.x * blockDim.x;
  for (long i = (long)blockIdx.x * blockDim.x + threadIdx.x; i < HEAD_ELEMS; i += stride) {
    int  d = (int)(i & 63);
    int  n = (int)((i >> 6) & 1023);
    long bh = i >> 16;
    int  h = (int)(bh % Hv);
    long b = bh / Hv;
    long src = (b * Nv + n) * (long)C3 + Cv + h * Dv + d;
    k16[i] = (_Float16)fqv(qkv[src], s, inv, z);
  }
}

// v16t[((b*H+h)*D+d)*N+n] = fq(qkv[(b*N+n)*3C + 2C + h*D + d])   (transposed)
__global__ __launch_bounds__(256)
void pack_v_kernel(const float* __restrict__ qkv, _Float16* __restrict__ v16t,
                   const QStat* __restrict__ st) {
  float s = st->scale, z = st->zp, inv = 1.0f / s;
  long stride = (long)gridDim.x * blockDim.x;
  for (long i = (long)blockIdx.x * blockDim.x + threadIdx.x; i < HEAD_ELEMS; i += stride) {
    int  n = (int)(i & 1023);
    int  d = (int)((i >> 10) & 63);
    long bh = i >> 16;
    int  h = (int)(bh % Hv);
    long b = bh / Hv;
    long src = (b * Nv + n) * (long)C3 + 2 * Cv + h * Dv + d;
    v16t[i] = (_Float16)fqv(qkv[src], s, inv, z);
  }
}

// ---------------------------------------------------------------------------
// Generic GEMM: C[m,n] = sum_k fq(A[m,k]) * B16[n,k]  (+bias[n])
// A: f32 (fake-quant fused at tile load), B: pre-quantized f16, K contiguous both.
// Block: 256 threads = 8 wave32, tile BM=128 x BN=64 x BK=32.
// Software-pipelined: stage tile ks+1 into the alternate LDS buffer while the
// WMMAs consume tile ks; one barrier per iteration.
// ---------------------------------------------------------------------------
#define BM 128
#define BN 64
#define BK 32
#define LDSA 36   // halfs per LDS row (32 + pad)
#define LDSB 36

#if defined(__HIP_DEVICE_COMPILE__) && __has_builtin(__builtin_amdgcn_global_load_async_to_lds_b128)
#define ASYNC_B 1
#else
#define ASYNC_B 0
#endif

union Frag { v16h v; unsigned u[8]; };

__global__ __launch_bounds__(256)
void gemm_aq_kernel(const float* __restrict__ A, long aOuter, long aInner,
                    const _Float16* __restrict__ Bm, long bOuter, long bInner,
                    float* __restrict__ C, long cOuter, long cInner,
                    int HH, int lda, int ldb, int ldc,
                    const float* __restrict__ bias,
                    const QStat* __restrict__ st,
                    int M, int N, int K) {
  __shared__ _Float16 As[2][BM * LDSA];
  __shared__ _Float16 Bs[2][BN * LDSB];

  int z = blockIdx.z;
  long zb = z / HH, zh = z % HH;
  const float*    Ab = A  + zb * aOuter + zh * aInner;
  const _Float16* Bb = Bm + zb * bOuter + zh * bInner;
  float*          Cb = C  + zb * cOuter + zh * cInner;

  int nBase = blockIdx.x * BN;
  int mBase = blockIdx.y * BM;

  int tid = threadIdx.x;
  int lane = tid & 31;
  int wave = tid >> 5;
  int laneHi = (lane >> 4) & 1;
  int lane16 = lane & 15;

  float s = st->scale, zp = st->zp;
  float inv = 1.0f / s;   // one division per thread, not per element

  v8f acc[4];
#pragma unroll
  for (int i = 0; i < 4; ++i) acc[i] = (v8f){0, 0, 0, 0, 0, 0, 0, 0};

  // tile-loader indices
  int aRow = tid >> 3;        // 0..31 (4 row-passes of 32)
  int aCol = (tid & 7) * 4;   // 0..28
  int bRow = tid >> 2;        // 0..63
  int bCol = (tid & 3) * 8;   // 0,8,16,24

  // --- staging helpers ---
  auto stageA = [&](int k0, int buf) {
#pragma unroll
    for (int i = 0; i < 4; ++i) {
      int r = aRow + i * 32;
      const float4 x4 = *(const float4*)(Ab + (long)(mBase + r) * lda + (k0 + aCol));
      unsigned u0 = packh2(fqv(x4.x, s, inv, zp), fqv(x4.y, s, inv, zp));
      unsigned u1 = packh2(fqv(x4.z, s, inv, zp), fqv(x4.w, s, inv, zp));
      *(uint2*)&As[buf][r * LDSA + aCol] = make_uint2(u0, u1);
    }
  };
  auto stageB = [&](int k0, int buf) {
    const _Float16* gp = Bb + (long)(nBase + bRow) * ldb + (k0 + bCol);
    _Float16* lp = &Bs[buf][bRow * LDSB + bCol];
#if ASYNC_B
    // CDNA5 async global->LDS DMA (tracked by ASYNCcnt), 16B per lane.
    __builtin_amdgcn_global_load_async_to_lds_b128((vi4*)gp, (vi4*)lp, 0, 0);
#else
    const uint4 d = *(const uint4*)gp;
    uint2* q = (uint2*)lp;
    q[0] = make_uint2(d.x, d.y);
    q[1] = make_uint2(d.z, d.w);
#endif
  };
  auto syncTiles = [&]() {
#if ASYNC_B
#if __has_builtin(__builtin_amdgcn_s_wait_asynccnt)
    __builtin_amdgcn_s_wait_asynccnt(0);
#else
    asm volatile("s_wait_asynccnt 0x0" ::: "memory");
#endif
#endif
    __syncthreads();
  };

  int nsteps = K / BK;

  // prologue: stage tile 0
  stageA(0, 0);
  stageB(0, 0);
  syncTiles();

  for (int ks = 0; ks < nsteps; ++ks) {
    int buf = ks & 1;

    // stage next tile into the alternate buffer while computing on this one
    if (ks + 1 < nsteps) {
      stageA((ks + 1) * BK, buf ^ 1);
      stageB((ks + 1) * BK, buf ^ 1);
      if (ks + 2 < nsteps) {   // hint the tile after next (global_prefetch_b8)
        __builtin_prefetch(Ab + (long)(mBase + aRow) * lda + ((ks + 2) * BK + aCol), 0, 0);
        __builtin_prefetch(Bb + (long)(nBase + bRow) * ldb + ((ks + 2) * BK + bCol), 0, 0);
      }
    }

    // --- A fragment (16x32 f16, ISA layout): lanes 0-15: M=lane, K pairs
    //     {0..7,16..23}; lanes 16-31: M=lane-16, K pairs {8..15,24..31} ---
    Frag a;
    {
      const unsigned* arow = (const unsigned*)&As[buf][(wave * 16 + lane16) * LDSA];
      int kb = laneHi * 4;  // dword offset
      a.u[0] = arow[kb + 0]; a.u[1] = arow[kb + 1];
      a.u[2] = arow[kb + 2]; a.u[3] = arow[kb + 3];
      a.u[4] = arow[kb + 8]; a.u[5] = arow[kb + 9];
      a.u[6] = arow[kb + 10]; a.u[7] = arow[kb + 11];
    }
#pragma unroll
    for (int ns = 0; ns < 4; ++ns) {
      // B fragment (32x16 f16): lane = N, lanes 0-15 K={0..7,16..23}, lanes 16-31 K={8..15,24..31}
      Frag b;
      const unsigned* brow = (const unsigned*)&Bs[buf][(ns * 16 + lane16) * LDSB];
      int kc = laneHi * 4;
#pragma unroll
      for (int j = 0; j < 8; ++j) b.u[j] = brow[kc + j + (j >= 4 ? 4 : 0)];
      acc[ns] = __builtin_amdgcn_wmma_f32_16x16x32_f16(
          false, a.v, false, b.v, (short)0, acc[ns], false, false);
    }

    // end of iteration: next-tile staging complete + this tile's readers done
    syncTiles();
  }

  // --- epilogue: C layout (8 VGPRs): VGPR r -> M = r + laneHi*8, N = lane16 ---
#pragma unroll
  for (int ns = 0; ns < 4; ++ns) {
    int n = nBase + ns * 16 + lane16;
    float bv = bias ? bias[n] : 0.0f;
#pragma unroll
    for (int r = 0; r < 8; ++r) {
      int m = mBase + wave * 16 + r + laneHi * 8;
      Cb[(long)m * ldc + n] = acc[ns][r] + bv;
    }
  }
}

// ---------------------------------------------------------------------------
// Softmax over rows of 1024 (in place, logits pre-scaled by `scale`).
// Also folds max(softmax(row)) = 1/rowsum into the attn stats slot.
// ---------------------------------------------------------------------------
__global__ __launch_bounds__(256)
void softmax_kernel(float* __restrict__ attn, QStat* st, float scale) {
  long row = blockIdx.x;
  float* p = attn + row * (long)Nv;
  int tid = threadIdx.x;
  float v[4];
  float m = -3.402823466e+38f;
#pragma unroll
  for (int j = 0; j < 4; ++j) {
    v[j] = p[tid + j * 256] * scale;
    m = fmaxf(m, v[j]);
  }
  __shared__ float red[256];
  red[tid] = m; __syncthreads();
  for (int s = 128; s > 0; s >>= 1) {
    if (tid < s) red[tid] = fmaxf(red[tid], red[tid + s]);
    __syncthreads();
  }
  m = red[0]; __syncthreads();
  float sum = 0.0f;
#pragma unroll
  for (int j = 0; j < 4; ++j) { v[j] = expf(v[j] - m); sum += v[j]; }
  red[tid] = sum; __syncthreads();
  for (int s = 128; s > 0; s >>= 1) {
    if (tid < s) red[tid] += red[tid + s];
    __syncthreads();
  }
  float inv = 1.0f / red[0];
#pragma unroll
  for (int j = 0; j < 4; ++j) p[tid + j * 256] = v[j] * inv;
  if (tid == 0) atomicMax(&st->encMax, encf(inv));  // row max of softmax = 1/sum
}

// ---------------------------------------------------------------------------
// Host launcher
// ---------------------------------------------------------------------------
extern "C" void kernel_launch(void* const* d_in, const int* in_sizes, int n_in,
                              void* d_out, int out_size, void* d_ws, size_t ws_size,
                              hipStream_t stream) {
  const float* x      = (const float*)d_in[0];  // [16,1024,768]
  const float* w_qkv  = (const float*)d_in[1];  // [2304,768]
  const float* b_qkv  = (const float*)d_in[2];  // [2304]
  const float* w_proj = (const float*)d_in[3];  // [768,768]
  const float* b_proj = (const float*)d_in[4];  // [768]

  float* out  = (float*)d_out;                  // [16,1024,768]
  float* attn = (float*)d_out + OUT_ELEMS;      // [16,12,1024,1024]

  // workspace carve-up (256B aligned)
  char* ws = (char*)d_ws;
  size_t cur = 0;
  auto alloc = [&](size_t bytes) {
    void* p = ws + cur;
    cur += (bytes + 255) & ~(size_t)255;
    return p;
  };
  QStat*    stats   = (QStat*)alloc(8 * sizeof(QStat));
  float*    qkv     = (float*)alloc((size_t)M1 * C3 * 4);          // 151 MB
  _Float16* wqkv16  = (_Float16*)alloc((size_t)C3 * Cv * 2);       // 3.5 MB
  _Float16* wproj16 = (_Float16*)alloc((size_t)Cv * Cv * 2);       // 1.2 MB
  _Float16* k16     = (_Float16*)alloc((size_t)HEAD_ELEMS * 2);    // 25 MB
  _Float16* v16t    = (_Float16*)alloc((size_t)HEAD_ELEMS * 2);    // 25 MB
  float*    ctx     = (float*)alloc((size_t)M1 * Cv * 4);          // 50 MB
  (void)ws_size;

  const dim3 blk(256);

  // 1) stats for x, w_qkv, w_proj
  hipLaunchKernelGGL(init_stats_kernel, dim3(1), dim3(32), 0, stream, stats);
  hipLaunchKernelGGL(reduce_minmax_kernel, dim3(2048), blk, 0, stream,
                     x, (long)M1, Cv, Cv, stats + 0);
  hipLaunchKernelGGL(reduce_minmax_kernel, dim3(512), blk, 0, stream,
                     w_qkv, (long)C3, Cv, Cv, stats + 1);
  hipLaunchKernelGGL(reduce_minmax_kernel, dim3(256), blk, 0, stream,
                     w_proj, (long)Cv, Cv, Cv, stats + 2);
  hipLaunchKernelGGL(finalize_stats_kernel, dim3(1), dim3(8), 0, stream, stats, 0x07u);

  // 2) quantize weights to f16
  hipLaunchKernelGGL(quantize_f16_kernel, dim3(1024), blk, 0, stream,
                     w_qkv, wqkv16, (long)C3 * Cv, stats + 1);
  hipLaunchKernelGGL(quantize_f16_kernel, dim3(1024), blk, 0, stream,
                     w_proj, wproj16, (long)Cv * Cv, stats + 2);

  // 3) qkv = fq(x) @ fq(w_qkv)^T + b_qkv   [16384 x 2304], K=768
  hipLaunchKernelGGL(gemm_aq_kernel, dim3(C3 / BN, M1 / BM, 1), blk, 0, stream,
                     x, 0L, 0L, wqkv16, 0L, 0L, qkv, 0L, 0L,
                     1, Cv, Cv, C3, b_qkv, stats + 0, M1, C3, Cv);

  // 4) stats for q, k, v (strided slices of qkv)
  hipLaunchKernelGGL(reduce_minmax_kernel, dim3(2048), blk, 0, stream,
                     qkv, (long)M1, Cv, C3, stats + 3);
  hipLaunchKernelGGL(reduce_minmax_kernel, dim3(2048), blk, 0, stream,
                     qkv + Cv, (long)M1, Cv, C3, stats + 4);
  hipLaunchKernelGGL(reduce_minmax_kernel, dim3(2048), blk, 0, stream,
                     qkv + 2 * Cv, (long)M1, Cv, C3, stats + 5);
  hipLaunchKernelGGL(finalize_stats_kernel, dim3(1), dim3(8), 0, stream, stats, 0x38u);

  // 5) pack fq(k) -> [B,H,N,D] f16 ; fq(v) transposed -> [B,H,D,N] f16
  hipLaunchKernelGGL(pack_k_kernel, dim3(2048), blk, 0, stream, qkv, k16, stats + 4);
  hipLaunchKernelGGL(pack_v_kernel, dim3(2048), blk, 0, stream, qkv, v16t, stats + 5);

  // 6) logits = fq(q) @ fq(k)^T  per (b,h)  [1024 x 1024], K=64 -> attn region
  hipLaunchKernelGGL(gemm_aq_kernel, dim3(Nv / BN, Nv / BM, Bv * Hv), blk, 0, stream,
                     qkv, (long)Nv * C3, (long)Dv,
                     k16, (long)Hv * Nv * Dv, (long)Nv * Dv,
                     attn, (long)Hv * Nv * Nv, (long)Nv * Nv,
                     Hv, C3, Dv, Nv, (const float*)nullptr, stats + 3, Nv, Nv, Dv);

  // 7) softmax(logits * D^-0.5) in place; collect attn max into slot 6
  hipLaunchKernelGGL(softmax_kernel, dim3(Bv * Hv * Nv), blk, 0, stream,
                     attn, stats + 6, 0.125f);
  hipLaunchKernelGGL(finalize_stats_kernel, dim3(1), dim3(8), 0, stream, stats, 0x40u);

  // 8) ctx = fq(attn) @ fq(v)  per (b,h)  [1024 x 64], K=1024 -> [B,N,C] layout
  hipLaunchKernelGGL(gemm_aq_kernel, dim3(Dv / BN, Nv / BM, Bv * Hv), blk, 0, stream,
                     attn, (long)Hv * Nv * Nv, (long)Nv * Nv,
                     v16t, (long)Hv * Dv * Nv, (long)Dv * Nv,
                     ctx, (long)Nv * Cv, (long)Dv,
                     Hv, Nv, Nv, Cv, (const float*)nullptr, stats + 6, Nv, Dv, Nv);

  // 9) stats for ctx
  hipLaunchKernelGGL(reduce_minmax_kernel, dim3(2048), blk, 0, stream,
                     ctx, (long)M1, Cv, Cv, stats + 7);
  hipLaunchKernelGGL(finalize_stats_kernel, dim3(1), dim3(8), 0, stream, stats, 0x80u);

  // 10) out = fq(ctx) @ fq(w_proj)^T + b_proj   [16384 x 768], K=768
  hipLaunchKernelGGL(gemm_aq_kernel, dim3(Cv / BN, M1 / BM, 1), blk, 0, stream,
                     ctx, 0L, 0L, wproj16, 0L, 0L, out, 0L, 0L,
                     1, Cv, Cv, Cv, b_proj, stats + 2, M1, Cv, Cv);

  (void)in_sizes; (void)n_in; (void)out_size;
}
